// BLSTM_Generator_76003741270327
// MI455X (gfx1250) — compile-verified, matrix operand order checked by reading, not run on hardware
//
#include <hip/hip_runtime.h>

typedef __attribute__((ext_vector_type(16))) __bf16 v16bf;
typedef __attribute__((ext_vector_type(8)))  __bf16 v8bf;
typedef __attribute__((ext_vector_type(8)))  float  v8f;

#define T_STEPS 4096
#define I_DIM   128
#define H_DIM   1024
#define O_DIM   128
#define K_DIM   1152   // I + H
#define KT_L    36     // K_DIM / 32
#define G4H     4096   // 4*H
#define NWG     32     // workgroups cooperating per LSTM

// K index owned by (lane, element e) inside a 16x16x32 bf16 WMMA fragment
// (mirrors the ISA 16-bit A/B VGPR striping: lanes<16 hold K 0-7,16-23;
//  lanes>=16 hold K 8-15,24-31).
__device__ __host__ __forceinline__ int kmap(int lane, int e) {
    return e + ((e & 8) ? 8 : 0) + ((lane & 16) ? 8 : 0);
}

__device__ __forceinline__ float sigf(float x) { return 1.0f / (1.0f + __expf(-x)); }

// ---------------------------------------------------------------------------
// Pack f32 weights into bf16 WMMA B-fragment order + zero the step counters.
// Packed block (nt,kt) is 512 contiguous bf16; element (lane,e) at lane*16+e
// holds W[nt*16 + lane%16][kt*32 + kmap(lane,e)].
// ---------------------------------------------------------------------------
__global__ void pack_kernel(const float* __restrict__ fW, const float* __restrict__ bW,
                            const float* __restrict__ dW, const float* __restrict__ oW,
                            __bf16* __restrict__ fWp, __bf16* __restrict__ bWp,
                            __bf16* __restrict__ dWp, __bf16* __restrict__ oWp,
                            int* __restrict__ cnts, int ncnt) {
    const long lstm_sz = (long)G4H * K_DIM;                // 4,718,592
    const long total   = 3 * lstm_sz + (long)O_DIM * H_DIM;
    for (long idx = (long)blockIdx.x * blockDim.x + threadIdx.x; idx < total;
         idx += (long)gridDim.x * blockDim.x) {
        long li = idx;
        const float* W; __bf16* Wp; int KT, K;
        if (li < lstm_sz)            { W = fW; Wp = fWp; KT = KT_L; K = K_DIM; }
        else if (li < 2 * lstm_sz)   { W = bW; Wp = bWp; li -= lstm_sz;     KT = KT_L; K = K_DIM; }
        else if (li < 3 * lstm_sz)   { W = dW; Wp = dWp; li -= 2 * lstm_sz; KT = KT_L; K = K_DIM; }
        else                         { W = oW; Wp = oWp; li -= 3 * lstm_sz; KT = 32;   K = H_DIM; }
        int blk  = (int)(li >> 9);
        int r    = (int)(li & 511);
        int lane = r >> 4, e = r & 15;
        int nt   = blk / KT, kt = blk - nt * KT;
        int n    = nt * 16 + (lane & 15);
        int k    = kt * 32 + kmap(lane, e);
        Wp[li] = (__bf16)W[(long)n * K + k];
    }
    for (int i = blockIdx.x * blockDim.x + threadIdx.x; i < ncnt; i += gridDim.x * blockDim.x)
        cnts[i] = 0;
}

// ---------------------------------------------------------------------------
// Persistent recurrent LSTM: 32 workgroups x 128 threads per LSTM.
// WG `wg` owns h[wg*32 .. wg*32+32). Wave w (=gate type) computes 2 N-tiles.
// Inter-step sync: per-step agent-scope atomic counter (arrive + spin).
// ---------------------------------------------------------------------------
__device__ void lstm_seq(const float* __restrict__ it, const __bf16* __restrict__ Wp,
                         const float* __restrict__ bias, float* __restrict__ hbuf,
                         int* __restrict__ cnt, float* __restrict__ hs,
                         const float* __restrict__ ctxf, const float* __restrict__ ctxb,
                         int reverse, int wg, __bf16* vsh, float* gsh) {
    const int tid = threadIdx.x;
    const int w   = tid >> 5;           // wave id == gate type (i,f,o,g)
    const int lid = tid & 31;
    const int nt0 = w * 64 + wg * 2;    // first of two 16-row gate tiles
    const int klsel = (lid & 16) ? 8 : 0;
    float cst = 0.0f;                   // cell state chunk lives in wave-0 regs

    for (int s = 0; s < T_STEPS; ++s) {
        const int t = reverse ? (T_STEPS - 1 - s) : s;

        // v = bf16(concat(x_t, h_prev)) staged in LDS
        for (int idx = tid; idx < K_DIM; idx += 128) {
            float val;
            if (idx < I_DIM)       val = it[(long)t * I_DIM + idx];
            else if (s == 0)       val = ctxf ? 0.5f * (ctxf[idx - I_DIM] + ctxb[idx - I_DIM]) : 0.0f;
            else                   val = hbuf[idx - I_DIM];
            vsh[idx] = (__bf16)val;
        }
        __syncthreads();

        v8f acc0 = {}; v8f acc1 = {};
        for (int kt = 0; kt < KT_L; ++kt) {
            const int klo = kt * 32 + klsel;
            v8bf alo = *(const v8bf*)&vsh[klo];
            v8bf ahi = *(const v8bf*)&vsh[klo + 16];
            v16bf a;
#pragma unroll
            for (int i2 = 0; i2 < 8; ++i2) { a[i2] = alo[i2]; a[i2 + 8] = ahi[i2]; }
            v16bf b0 = *(const v16bf*)(Wp + ((long)(nt0 * KT_L + kt) << 9) + lid * 16);
            v16bf b1 = *(const v16bf*)(Wp + ((long)((nt0 + 1) * KT_L + kt) << 9) + lid * 16);
            acc0 = __builtin_amdgcn_wmma_f32_16x16x32_bf16(false, a, false, b0, (short)0, acc0, false, false);
            acc1 = __builtin_amdgcn_wmma_f32_16x16x32_bf16(false, a, false, b1, (short)0, acc1, false, false);
        }
        if (lid < 16) {   // every D row equals row 0 (A rows identical)
            gsh[w * 32 + lid]      = acc0[0] + bias[nt0 * 16 + lid];
            gsh[w * 32 + 16 + lid] = acc1[0] + bias[(nt0 + 1) * 16 + lid];
        }
        __syncthreads();

        if (tid < 32) {   // LSTM cell for our 32 h values
            float gi = sigf(gsh[tid]);
            float gf = sigf(gsh[32 + tid]);
            float go = sigf(gsh[64 + tid]);
            float gg = tanhf(gsh[96 + tid]);
            cst = gf * cst + gi * gg;
            float h = go * tanhf(cst);
            hbuf[wg * 32 + tid] = h;
            if (hs) hs[(long)t * H_DIM + wg * 32 + tid] = h;
        }
        __threadfence();
        __syncthreads();
        if (tid == 0) {
            __hip_atomic_fetch_add(&cnt[s], 1, __ATOMIC_RELEASE, __HIP_MEMORY_SCOPE_AGENT);
            while (__hip_atomic_load(&cnt[s], __ATOMIC_ACQUIRE, __HIP_MEMORY_SCOPE_AGENT) < NWG)
                __builtin_amdgcn_s_sleep(1);
        }
        __syncthreads();
        __threadfence();
    }
}

__global__ void __launch_bounds__(128)
enc_kernel(const float* it, const __bf16* fWp, const float* fb,
           const __bf16* bWp, const float* bb,
           float* hf, float* hb, int* cf, int* cb) {
    __shared__ __align__(16) __bf16 vsh[K_DIM];
    __shared__ float gsh[128];
    const int dir = blockIdx.x >> 5, wg = blockIdx.x & 31;
    if (dir == 0)
        lstm_seq(it, fWp, fb, hf, cf, nullptr, nullptr, nullptr, 0, wg, vsh, gsh);
    else
        lstm_seq(it, bWp, bb, hb, cb, nullptr, nullptr, nullptr, 1, wg, vsh, gsh);
}

__global__ void __launch_bounds__(128)
dec_kernel(const float* it, const __bf16* dWp, const float* db,
           const float* hf, const float* hb, float* hd, float* hs, int* cd) {
    __shared__ __align__(16) __bf16 vsh[K_DIM];
    __shared__ float gsh[128];
    lstm_seq(it, dWp, db, hd, cd, hs, hf, hb, 0, blockIdx.x, vsh, gsh);
}

// ---------------------------------------------------------------------------
// out = hs @ out_W^T + out_b : [4096,1024] x [1024,128]. 256 blocks (M-tiles),
// 8 waves each (one 16-col N-tile per wave), K swept in 32-wide WMMA steps.
// ---------------------------------------------------------------------------
__global__ void __launch_bounds__(256)
outgemm_kernel(const float* __restrict__ hs, const __bf16* __restrict__ oWp,
               const float* __restrict__ ob, float* __restrict__ out) {
    const int mt  = blockIdx.x;         // 0..255
    const int w   = threadIdx.x >> 5;   // N-tile 0..7
    const int lid = threadIdx.x & 31;
    const int m0  = mt * 16 + (lid & 15);
    const int klsel = (lid & 16) ? 8 : 0;
    const float* arow = hs + (long)m0 * H_DIM;
    v8f acc = {};
    for (int kt = 0; kt < 32; ++kt) {
        const int klo = kt * 32 + klsel;
        v16bf a;
#pragma unroll
        for (int i2 = 0; i2 < 8; ++i2) {
            a[i2]     = (__bf16)arow[klo + i2];
            a[i2 + 8] = (__bf16)arow[klo + 16 + i2];
        }
        v16bf b = *(const v16bf*)(oWp + ((long)(w * 32 + kt) << 9) + lid * 16);
        acc = __builtin_amdgcn_wmma_f32_16x16x32_bf16(false, a, false, b, (short)0, acc, false, false);
    }
    const int n  = w * 16 + (lid & 15);
    const float bn = ob[n];
    const int mbase = mt * 16 + ((lid & 16) ? 8 : 0);
#pragma unroll
    for (int r = 0; r < 8; ++r)
        out[(long)(mbase + r) * O_DIM + n] = acc[r] + bn;
}

extern "C" void kernel_launch(void* const* d_in, const int* in_sizes, int n_in,
                              void* d_out, int out_size, void* d_ws, size_t ws_size,
                              hipStream_t stream) {
    (void)in_sizes; (void)n_in; (void)out_size; (void)ws_size;
    const float* it = (const float*)d_in[0];
    const float* fW = (const float*)d_in[1];
    const float* fb = (const float*)d_in[2];
    const float* bW = (const float*)d_in[3];
    const float* bb = (const float*)d_in[4];
    const float* dW = (const float*)d_in[5];
    const float* db = (const float*)d_in[6];
    const float* oW = (const float*)d_in[7];
    const float* ob = (const float*)d_in[8];

    char* ws = (char*)d_ws;
    size_t off = 0;
    auto carve = [&](size_t bytes) -> void* {
        void* p = ws + off;
        off = (off + bytes + 255) & ~(size_t)255;
        return p;
    };
    const size_t lstm_sz = (size_t)G4H * K_DIM;             // elements
    __bf16* fWp = (__bf16*)carve(lstm_sz * 2);
    __bf16* bWp = (__bf16*)carve(lstm_sz * 2);
    __bf16* dWp = (__bf16*)carve(lstm_sz * 2);
    __bf16* oWp = (__bf16*)carve((size_t)O_DIM * H_DIM * 2);
    float*  hf  = (float*)carve(H_DIM * 4);
    float*  hb  = (float*)carve(H_DIM * 4);
    float*  hd  = (float*)carve(H_DIM * 4);
    float*  hs  = (float*)carve((size_t)T_STEPS * H_DIM * 4);
    int*    cnts = (int*)carve(3 * T_STEPS * 4);
    int* cf = cnts;
    int* cb = cnts + T_STEPS;
    int* cd = cnts + 2 * T_STEPS;

    pack_kernel<<<4096, 256, 0, stream>>>(fW, bW, dW, oW, fWp, bWp, dWp, oWp,
                                          cnts, 3 * T_STEPS);
    enc_kernel<<<2 * NWG, 128, 0, stream>>>(it, fWp, fb, bWp, bb, hf, hb, cf, cb);
    dec_kernel<<<NWG, 128, 0, stream>>>(it, dWp, db, hf, hb, hd, hs, cd);
    outgemm_kernel<<<T_STEPS / 16, 256, 0, stream>>>(hs, oWp, ob, (float*)d_out);
}
